// MessageBlock_38929583571221
// MI455X (gfx1250) — compile-verified, matrix operand order checked by reading, not run on hardware
//
#include <hip/hip_runtime.h>

#define F    128
#define F3   384
#define TE   32     // edges per block (two 16-row M-tiles)
#define TN   16     // nodes per block
#define NRBF 20

typedef __bf16 bf16_t;
typedef __attribute__((ext_vector_type(16))) __bf16 v16bf;
typedef __attribute__((ext_vector_type(8)))  float  v8f;

// ---------------------------------------------------------------------------
// WMMA fragment order (CDNA5 ISA 7.12.2, wave32, 16-bit A 16x32):
//   lane = 16*(k bit3) + row ;  r = 4*(k bit4) + (k bits2:1) ; pair elem = k bit0
// We store A-tiles in LDS and B-tiles (weights) in global memory directly in
// this order, so a fragment load is one contiguous 32-byte per-lane read
// (2 x b128) with no half-register repacking.
// Packed block layout: [K/32][32 lanes][16 bf16]  (B adds an N/16 tile dim).
// ---------------------------------------------------------------------------
static __device__ __forceinline__ int a_pk_idx(int row, int k) {
    int kb = k >> 5, kl = k & 31;
    int r    = ((kl >> 4) & 1) * 4 + ((kl >> 1) & 3);
    int lane = ((kl >> 3) & 1) * 16 + row;
    return ((kb * 32 + lane) << 4) + (r << 1) + (kl & 1);
}

static __device__ __forceinline__ v16bf load_frag(const bf16_t* __restrict__ p, int blk32) {
    return *(const v16bf*)(p + (blk32 << 4));   // 32B-aligned contiguous read
}

static __device__ __forceinline__ v8f wmma_bf16(v16bf a, v16bf b, v8f c) {
    return __builtin_amdgcn_wmma_f32_16x16x32_bf16(false, a, false, b, (short)0, c,
                                                   false, false);
}

static __device__ __forceinline__ float fast_silu(float x) {
    return x * __builtin_amdgcn_rcpf(1.0f + __expf(-x));   // v_exp + v_rcp, no IEEE div
}

// ---------------- prep: convert + pad + pack B matrices into fragment order --
__global__ void pack_b_kernel(const float* __restrict__ src, bf16_t* __restrict__ dst,
                              int Ksrc, int Kpad, int N) {
    int dw = blockIdx.x * blockDim.x + threadIdx.x;   // one bf16 pair per thread
    int total = (Kpad >> 1) * N;
    if (dw >= total) return;
    int r    = dw & 7;
    int lane = (dw >> 3) & 31;
    int tile = dw >> 8;                               // kb * NT + nt
    int NT   = N >> 4;
    int nt   = tile % NT;
    int kb   = tile / NT;
    int k    = kb * 32 + ((r & 4) << 2) + ((lane >> 4) << 3) + ((r & 3) << 1);
    int n    = nt * 16 + (lane & 15);
    float x0 = (k     < Ksrc) ? src[k * N + n]       : 0.0f;
    float x1 = (k + 1 < Ksrc) ? src[(k + 1) * N + n] : 0.0f;
    dst[2 * dw]     = (bf16_t)x0;
    dst[2 * dw + 1] = (bf16_t)x1;
}

__global__ void init_out_kernel(const float* __restrict__ s, const float* __restrict__ v,
                                float* __restrict__ out, int ns, int nv) {
    int i = blockIdx.x * blockDim.x + threadIdx.x;
    if (i < ns)           out[i] = s[i];
    else if (i < ns + nv) out[i] = v[i - ns];
}

// ---------------- edge (message) kernel ----------------
// block = 8 waves, 32 edges (2 M-tiles share every B fragment).
__global__ __launch_bounds__(256) void edge_message_kernel(
    const float* __restrict__ s, const float* __restrict__ v,
    const float* __restrict__ rbf, const float* __restrict__ dir,
    const float* __restrict__ b1, const float* __restrict__ b2,
    const float* __restrict__ br,
    const bf16_t* __restrict__ W1pk, const bf16_t* __restrict__ W2pk,
    const bf16_t* __restrict__ Wrpk,
    const int* __restrict__ i_index, const int* __restrict__ j_index,
    float* __restrict__ s_out, float* __restrict__ v_out, int E_)
{
    __shared__ __align__(32) bf16_t sApk[2 * 4 * 32 * 16];   // 8 KB, A tiles; reused for hidden
    __shared__ __align__(32) bf16_t rbfApk[2 * 1 * 32 * 16]; // 2 KB, K padded 20->32
    __shared__ float  splitS[TE * F3];                        // 48 KB
    __shared__ float  dirS[TE * 3];
    __shared__ int    jS[TE], iS[TE];

    int e0 = blockIdx.x * TE;
    int t  = threadIdx.x;
    if (t < TE) {
        int e = min(e0 + t, E_ - 1);
        jS[t] = j_index[e];
        iS[t] = i_index[e];
    }
    if (t < TE * 3) {
        int e  = t / 3, d = t % 3;
        int ge = min(e0 + e, E_ - 1);
        dirS[t] = dir[(size_t)ge * 3 + d];
    }
    __syncthreads();

    // gather s[j] straight into packed-A order (pair-wise writes)
    for (int idx = t; idx < TE * (F / 2); idx += 256) {
        int row = idx >> 6;            // edge within tile 0..31
        int kp  = (idx & 63) << 1;
        const float* sp = s + (size_t)jS[row] * F + kp;
        int base = (row >> 4) * 2048 + a_pk_idx(row & 15, kp);
        sApk[base]     = (bf16_t)sp[0];
        sApk[base + 1] = (bf16_t)sp[1];
    }
    for (int idx = t; idx < TE * 16; idx += 256) {
        int row = idx >> 4;
        int kp  = (idx & 15) << 1;
        int ge  = min(e0 + row, E_ - 1);
        float x0 = (kp     < NRBF) ? rbf[(size_t)ge * NRBF + kp]     : 0.0f;
        float x1 = (kp + 1 < NRBF) ? rbf[(size_t)ge * NRBF + kp + 1] : 0.0f;
        int base = (row >> 4) * 512 + a_pk_idx(row & 15, kp);
        rbfApk[base]     = (bf16_t)x0;
        rbfApk[base + 1] = (bf16_t)x1;
    }
    __syncthreads();

    int wv = t >> 5, lane = t & 31;
    int col = lane & 15, rowhi = (lane >> 4) << 3;
    int g0 = wv * 16;

    // hidden = silu(s[j] @ W1 + b1); keep in regs across the barrier
    float h0[8], h1[8];
    {
        v8f acc0 = {}, acc1 = {};
#pragma unroll
        for (int kb = 0; kb < 4; ++kb) {
            if (kb < 3)
                __builtin_prefetch(W1pk + (((kb + 1) * 8 + wv) * 32 + lane) * 16, 0, 3);
            v16bf b  = load_frag(W1pk, (kb * 8 + wv) * 32 + lane);
            v16bf a0 = load_frag(sApk,        kb * 32 + lane);
            v16bf a1 = load_frag(sApk + 2048, kb * 32 + lane);
            acc0 = wmma_bf16(a0, b, acc0);
            acc1 = wmma_bf16(a1, b, acc1);
        }
        float bias = b1[g0 + col];
#pragma unroll
        for (int r = 0; r < 8; ++r) {
            h0[r] = fast_silu(acc0[r] + bias);
            h1[r] = fast_silu(acc1[r] + bias);
        }
    }
    __syncthreads();          // everyone done reading sApk
    {
        int k = g0 + col;     // hidden feature index this lane produced
#pragma unroll
        for (int r = 0; r < 8; ++r) {
            int row = r + rowhi;
            sApk[a_pk_idx(row, k)]        = (bf16_t)h0[r];
            sApk[2048 + a_pk_idx(row, k)] = (bf16_t)h1[r];
        }
    }
    __syncthreads();

    // phi = hidden @ W2 + b2 ; x = rbf @ Wr + br ; split = phi * cutoff(x)
#pragma unroll
    for (int j = 0; j < 3; ++j) {
        int nt = wv + 8 * j;
        int c0 = nt * 16;
        v8f p0 = {}, p1 = {};
#pragma unroll
        for (int kb = 0; kb < 4; ++kb) {
            v16bf b  = load_frag(W2pk, (kb * 24 + nt) * 32 + lane);
            v16bf a0 = load_frag(sApk,        kb * 32 + lane);
            v16bf a1 = load_frag(sApk + 2048, kb * 32 + lane);
            p0 = wmma_bf16(a0, b, p0);
            p1 = wmma_bf16(a1, b, p1);
        }
        v8f x0 = {}, x1 = {};
        {
            v16bf b  = load_frag(Wrpk, nt * 32 + lane);
            v16bf a0 = load_frag(rbfApk,       lane);
            v16bf a1 = load_frag(rbfApk + 512, lane);
            x0 = wmma_bf16(a0, b, x0);
            x1 = wmma_bf16(a1, b, x1);
        }
        float bias2 = b2[c0 + col], biasr = br[c0 + col];
#pragma unroll
        for (int r = 0; r < 8; ++r) {
            int row = r + rowhi;
            float xx0 = x0[r] + biasr, xx1 = x1[r] + biasr;
            float W0 = (xx0 < 5.0f) ? 0.5f * (__cosf(xx0 * 0.6283185307179586f) + 1.0f) : 0.0f;
            float W1v = (xx1 < 5.0f) ? 0.5f * (__cosf(xx1 * 0.6283185307179586f) + 1.0f) : 0.0f;
            splitS[row * F3 + c0 + col]        = (p0[r] + bias2) * W0;
            splitS[(16 + row) * F3 + c0 + col] = (p1[r] + bias2) * W1v;
        }
    }
    __syncthreads();

    // scatter: s_out[i] += Ws ; v_out[i] += Wvs*dir + Wvv*v[j]   (L2-resident atomics)
    for (int idx = t; idx < TE * F; idx += 256) {
        int e = idx >> 7, f = idx & 127;
        if (e0 + e >= E_) continue;
        atomicAdd(&s_out[(size_t)iS[e] * F + f], splitS[e * F3 + f]);
        float wvs = splitS[e * F3 + F + f];
        float wvv = splitS[e * F3 + 2 * F + f];
        const float* vj = v     + (size_t)jS[e] * F3 + f * 3;
        float*       vo = v_out + (size_t)iS[e] * F3 + f * 3;
#pragma unroll
        for (int d = 0; d < 3; ++d)
            atomicAdd(&vo[d], wvs * dirS[e * 3 + d] + wvv * vj[d]);
    }
}

// ---------------- node (update) kernel ----------------
__global__ __launch_bounds__(256) void node_update_kernel(
    const bf16_t* __restrict__ WUpk, const bf16_t* __restrict__ WVpk,
    const bf16_t* __restrict__ M1pk, const bf16_t* __restrict__ M2pk,
    const float* __restrict__ bU, const float* __restrict__ bV,
    const float* __restrict__ bm1, const float* __restrict__ bm2,
    float* __restrict__ s_io, float* __restrict__ v_io, int N_)
{
    // phase-overlapped scratch: vAp (12KB) -> mlpA (8KB) -> mS (12KB)
    __shared__ __align__(32) char uScr[3 * 4 * 32 * 16 * sizeof(bf16_t)];
    __shared__ float  UvS[3][TN * F];                 // 24 KB, f32
    __shared__ bf16_t VvS[3][TN * F];                 // 12 KB
    __shared__ __align__(32) bf16_t hidp[4 * 32 * 16]; // 4 KB, packed-A hidden

    int n0 = blockIdx.x * TN;
    int t  = threadIdx.x;
    int wv = t >> 5, lane = t & 31;
    int col = lane & 15, rowhi = (lane >> 4) << 3;
    int g0 = wv * 16;

    // P1: v' tile -> three packed-A matrices (one per xyz component)
    bf16_t* vAp = (bf16_t*)uScr;
    for (int idx = t; idx < TN * (F / 2); idx += 256) {
        int n  = idx >> 6;
        int kp = (idx & 63) << 1;
        int gn = min(n0 + n, N_ - 1);
        const float* vp = v_io + (size_t)gn * F3 + kp * 3;
        int base = a_pk_idx(n, kp);
#pragma unroll
        for (int d = 0; d < 3; ++d) {
            vAp[d * 2048 + base]     = (bf16_t)vp[d];
            vAp[d * 2048 + base + 1] = (bf16_t)vp[3 + d];
        }
    }
    __syncthreads();

    // P2: Uv = v'@WU + bU ; Vv = v'@WV + bV
#pragma unroll
    for (int d = 0; d < 3; ++d) {
        v8f accU = {}, accV = {};
#pragma unroll
        for (int kb = 0; kb < 4; ++kb) {
            v16bf a   = load_frag(vAp + d * 2048, kb * 32 + lane);
            v16bf bu_ = load_frag(WUpk, (kb * 8 + wv) * 32 + lane);
            v16bf bv_ = load_frag(WVpk, (kb * 8 + wv) * 32 + lane);
            accU = wmma_bf16(a, bu_, accU);
            accV = wmma_bf16(a, bv_, accV);
        }
        float biasU = bU[g0 + col], biasV = bV[g0 + col];
#pragma unroll
        for (int r = 0; r < 8; ++r) {
            int row = r + rowhi;
            UvS[d][row * F + g0 + col] = accU[r] + biasU;
            VvS[d][row * F + g0 + col] = (bf16_t)(accV[r] + biasV);
        }
    }
    __syncthreads();

    // P3: mlp_in = [ |Vv| , s' ]  packed as 16x256 A-matrix
    bf16_t* mlpA = (bf16_t*)uScr;
    for (int idx = t; idx < TN * F; idx += 256) {
        int n = idx >> 7, g = idx & 127;
        float nrm = 0.f;
#pragma unroll
        for (int d = 0; d < 3; ++d) { float x = (float)VvS[d][idx]; nrm += x * x; }
        int gn = min(n0 + n, N_ - 1);
        mlpA[a_pk_idx(n, g)]     = (bf16_t)sqrtf(nrm);
        mlpA[a_pk_idx(n, F + g)] = (bf16_t)s_io[(size_t)gn * F + g];
    }
    __syncthreads();

    // P4: hidden = silu(mlp_in @ M1 + bm1)
    {
        v8f acc = {};
#pragma unroll
        for (int kb = 0; kb < 8; ++kb) {
            v16bf a = load_frag(mlpA, kb * 32 + lane);
            v16bf b = load_frag(M1pk, (kb * 8 + wv) * 32 + lane);
            acc = wmma_bf16(a, b, acc);
        }
        float bias = bm1[g0 + col];
        int k = g0 + col;
#pragma unroll
        for (int r = 0; r < 8; ++r)
            hidp[a_pk_idx(r + rowhi, k)] = (bf16_t)fast_silu(acc[r] + bias);
    }
    __syncthreads();

    // P5: m = hidden @ M2 + bm2 -> [a_vv | a_sv | a_ss]
    bf16_t* mS = (bf16_t*)uScr;            // mlpA dead
#pragma unroll
    for (int j = 0; j < 3; ++j) {
        int nt = wv + 8 * j;
        int c0 = nt * 16;
        v8f acc = {};
#pragma unroll
        for (int kb = 0; kb < 4; ++kb) {
            v16bf a = load_frag(hidp, kb * 32 + lane);
            v16bf b = load_frag(M2pk, (kb * 24 + nt) * 32 + lane);
            acc = wmma_bf16(a, b, acc);
        }
        float bias = bm2[c0 + col];
#pragma unroll
        for (int r = 0; r < 8; ++r)
            mS[(r + rowhi) * F3 + c0 + col] = (bf16_t)(acc[r] + bias);
    }
    __syncthreads();

    // P6: s += (Uv.Vv)*a_sv + a_ss ; v += a_vv*Uv   (in place)
    for (int idx = t; idx < TN * F; idx += 256) {
        int n = idx >> 7, g = idx & 127;
        int gn = n0 + n;
        if (gn >= N_) continue;
        float a_vv = (float)mS[n * F3 + g];
        float a_sv = (float)mS[n * F3 + F + g];
        float a_ss = (float)mS[n * F3 + 2 * F + g];
        float dot = 0.f;
        float* vp = v_io + (size_t)gn * F3 + g * 3;
#pragma unroll
        for (int d = 0; d < 3; ++d) {
            float uvd = UvS[d][idx];
            dot  += uvd * (float)VvS[d][idx];
            vp[d] = vp[d] + a_vv * uvd;
        }
        size_t si = (size_t)gn * F + g;
        s_io[si] = s_io[si] + dot * a_sv + a_ss;
    }
}

// ---------------- launcher ----------------
extern "C" void kernel_launch(void* const* d_in, const int* in_sizes, int n_in,
                              void* d_out, int out_size, void* d_ws, size_t ws_size,
                              hipStream_t stream) {
    const float* s   = (const float*)d_in[0];
    const float* v   = (const float*)d_in[1];
    const float* rbf = (const float*)d_in[2];
    const float* dir = (const float*)d_in[3];
    const float* W1  = (const float*)d_in[4];
    const float* b1  = (const float*)d_in[5];
    const float* W2  = (const float*)d_in[6];
    const float* b2  = (const float*)d_in[7];
    const float* Wr  = (const float*)d_in[8];
    const float* br  = (const float*)d_in[9];
    const float* WU  = (const float*)d_in[10];
    const float* bU  = (const float*)d_in[11];
    const float* WV  = (const float*)d_in[12];
    const float* bV  = (const float*)d_in[13];
    const float* M1  = (const float*)d_in[14];
    const float* bm1 = (const float*)d_in[15];
    const float* M2  = (const float*)d_in[16];
    const float* bm2 = (const float*)d_in[17];
    const int* i_index = (const int*)d_in[18];
    const int* j_index = (const int*)d_in[19];

    int N_ = in_sizes[0] / F;
    int E_ = in_sizes[18];
    int ns = N_ * F, nv = N_ * F * 3;

    // fragment-packed bf16 weights in workspace (~385 KB)
    bf16_t* W1pk = (bf16_t*)d_ws;
    bf16_t* W2pk = W1pk + F * F;
    bf16_t* WUpk = W2pk + F * F3;
    bf16_t* WVpk = WUpk + F * F;
    bf16_t* M1pk = WVpk + F * F;
    bf16_t* M2pk = M1pk + 2 * F * F;
    bf16_t* Wrpk = M2pk + F * F3;

    auto blk = [](int n) { return (n + 255) / 256; };
    pack_b_kernel<<<blk((F / 2) * F),      256, 0, stream>>>(W1, W1pk, F, F, F);
    pack_b_kernel<<<blk((F / 2) * F3),     256, 0, stream>>>(W2, W2pk, F, F, F3);
    pack_b_kernel<<<blk((F / 2) * F),      256, 0, stream>>>(WU, WUpk, F, F, F);
    pack_b_kernel<<<blk((F / 2) * F),      256, 0, stream>>>(WV, WVpk, F, F, F);
    pack_b_kernel<<<blk(F * F),            256, 0, stream>>>(M1, M1pk, 2 * F, 2 * F, F);
    pack_b_kernel<<<blk((F / 2) * F3),     256, 0, stream>>>(M2, M2pk, F, F, F3);
    pack_b_kernel<<<blk(16 * F3),          256, 0, stream>>>(Wr, Wrpk, NRBF, 32, F3);

    float* s_out = (float*)d_out;
    float* v_out = s_out + ns;
    init_out_kernel<<<blk(ns + nv), 256, 0, stream>>>(s, v, (float*)d_out, ns, nv);

    edge_message_kernel<<<(E_ + TE - 1) / TE, 256, 0, stream>>>(
        s, v, rbf, dir, b1, b2, br, W1pk, W2pk, Wrpk, i_index, j_index, s_out, v_out, E_);

    node_update_kernel<<<(N_ + TN - 1) / TN, 256, 0, stream>>>(
        WUpk, WVpk, M1pk, M2pk, bU, bV, bm1, bm2, s_out, v_out, N_);
}
